// LevelDepthAwareCrossAttention_48215302865356
// MI455X (gfx1250) — compile-verified, matrix operand order checked by reading, not run on hardware
//
#include <hip/hip_runtime.h>
#include <math.h>

// ---------------------------------------------------------------------------
// CDNA5 (gfx1250) implementation. wave32, WMMA f16 16x16x32 with f32 accum.
// A-fragments loaded straight from global as 2x b128 per lane (the 16-bit
// A 16x32 layout is two contiguous 8-half runs per lane). B (weights) are
// pre-permuted into fragment order once per level, so B-fragments are one
// contiguous 32B run per lane. Each wave computes 32x64 output per k-step
// (2 A-frags share 4 B-frags -> 8 WMMAs), fragments preloaded as a batch so
// loads overlap the WMMA burst. Attention K/V staged via async global->LDS.
// ---------------------------------------------------------------------------

typedef __attribute__((ext_vector_type(16))) _Float16 v16h;
typedef __attribute__((ext_vector_type(8)))  _Float16 v8h;
typedef __attribute__((ext_vector_type(8)))  float    v8f;

#define NHEADS 8

__device__ __forceinline__ v16h cat8(v8h lo, v8h hi) {
  return __builtin_shufflevector(lo, hi, 0, 1, 2, 3, 4, 5, 6, 7,
                                 8, 9, 10, 11, 12, 13, 14, 15);
}

// Fragment-order index for a KxN f16 B matrix (WMMA B 32x16 16-bit layout).
// Storage: [K/32][N/16][lane:32][e:16] halfs.
__device__ __forceinline__ size_t frag_index(int k, int n, int N) {
  int kb = k >> 5, kr = k & 31;
  int hi = (kr >> 3) & 1;
  int g  = (((kr >> 4) & 1) << 2) | ((kr >> 1) & 3);
  int e  = (g << 1) | (kr & 1);
  int nb = n >> 4, nlo = n & 15;
  int lane = (hi << 4) | nlo;
  return (((size_t)kb * (N >> 4) + nb) * 32 + lane) * (size_t)16 + e;
}

// async global->LDS b128 (CDNA5 ISA); ldsOff is the
// wave-relative LDS byte offset (low 32 bits of the generic shared address).
__device__ __forceinline__ void async_g2l_b128(unsigned ldsOff, const void* g) {
  asm volatile("global_load_async_to_lds_b128 %0, %1, off"
               :: "v"(ldsOff), "v"((unsigned long long)g) : "memory");
}
__device__ __forceinline__ void wait_async0() {
  asm volatile("s_wait_asynccnt 0" ::: "memory");
}

// Polar grid coordinates (matches _polar_coords in the reference).
__device__ __forceinline__ void polar_xy(int i, int w, int H, int W, int h1, int w2,
                                         float r00, float r10, float fx, float cx,
                                         float& X, float& Y) {
  float cr = r10;          // R = [[0,1],[-1,0]] @ rot[:2,:2] -> R[0,0]=rot[1,0]
  float sr = -r00;         //                                  R[1,0]=-rot[0,0]
  float pw   = 800.0f / (float)w2;
  float orig = (float)w * pw + 0.5f * pw;
  float ang  = atanf((orig - cx) / fx);
  float c_ = cosf(ang), s_ = sinf(ang);
  float ca = cr * c_ + sr * s_;
  float sa = -sr * c_ + cr * s_;
  float cx0 = (float)(W / 2), cy0 = (float)(H / 2);
  float rmax = sqrtf(cx0 * cx0 + cy0 * cy0);
  float rad  = ((float)i / (float)(h1 - 1)) * rmax;
  X = cx0 + rad * ca;
  Y = cy0 - rad * sa;
}

// ---------------------------------------------------------------------------
__global__ void copy_f32_kernel(const float* __restrict__ src, float* __restrict__ dst,
                                size_t n) {
  size_t i = (size_t)blockIdx.x * blockDim.x + threadIdx.x;
  size_t stride = (size_t)gridDim.x * blockDim.x;
  for (; i < n; i += stride) dst[i] = src[i];
}

// ---------------------------------------------------------------------------
// Prefuse MHA in-projection with the q/k/v projections and emit the combined
// weights directly in WMMA-fragment order (f16). blockIdx.y: q/k/v/out.
// ---------------------------------------------------------------------------
__global__ void combine_weights_kernel(const float* __restrict__ wq, const float* __restrict__ bq,
                                       const float* __restrict__ wk, const float* __restrict__ bk,
                                       const float* __restrict__ wv, const float* __restrict__ bv,
                                       const float* __restrict__ win, const float* __restrict__ bin,
                                       const float* __restrict__ wout, const float* __restrict__ bout,
                                       _Float16* WqF, _Float16* WkF, _Float16* WvF, _Float16* WoF,
                                       float* bQ, float* bK, float* bV, float* bO,
                                       int d, int c2) {
  int which = blockIdx.y;
  int idx = blockIdx.x * blockDim.x + threadIdx.x;
  float scale = rsqrtf((float)(d / NHEADS));
  if (which == 0) {                       // q: K=d, N=d
    if (idx >= d * d) return;
    int k = idx / d, j = idx % d;
    float s = 0.f;
    for (int t = 0; t < d; ++t) s += win[j * d + t] * wq[t * d + k];
    WqF[frag_index(k, j, d)] = (_Float16)(s * scale);
    if (k == 0) {
      float sb = 0.f;
      for (int t = 0; t < d; ++t) sb += win[j * d + t] * bq[t];
      bQ[j] = (sb + bin[j]) * scale;
    }
  } else if (which == 1) {                // k: K=c2, N=d
    if (idx >= c2 * d) return;
    int k = idx / d, j = idx % d;
    const float* wkin = win + (size_t)d * d;
    float s = 0.f;
    for (int t = 0; t < d; ++t) s += wkin[j * d + t] * wk[t * c2 + k];
    WkF[frag_index(k, j, d)] = (_Float16)s;
    if (k == 0) {
      float sb = 0.f;
      for (int t = 0; t < d; ++t) sb += wkin[j * d + t] * bk[t];
      bK[j] = sb + bin[d + j];
    }
  } else if (which == 2) {                // v: K=c2, N=d
    if (idx >= c2 * d) return;
    int k = idx / d, j = idx % d;
    const float* wvin = win + (size_t)2 * d * d;
    float s = 0.f;
    for (int t = 0; t < d; ++t) s += wvin[j * d + t] * wv[t * c2 + k];
    WvF[frag_index(k, j, d)] = (_Float16)s;
    if (k == 0) {
      float sb = 0.f;
      for (int t = 0; t < d; ++t) sb += wvin[j * d + t] * bv[t];
      bV[j] = sb + bin[2 * d + j];
    }
  } else {                                // out-proj: K=d, N=d
    if (idx >= d * d) return;
    int k = idx / d, j = idx % d;
    WoF[frag_index(k, j, d)] = (_Float16)wout[j * d + k];
    if (k == 0) bO[j] = bout[j];
  }
}

// ---------------------------------------------------------------------------
// Extract: polar bilinear gather of working a (n,C,H,W) -> a_seq f16
// ---------------------------------------------------------------------------
__global__ void extract_seq_kernel(const float* __restrict__ aWork,
                                   const float* __restrict__ rots,
                                   const float* __restrict__ fxs,
                                   const float* __restrict__ cxs,
                                   const float* __restrict__ pa,
                                   _Float16* __restrict__ aseq,
                                   int C, int H, int W, int h1, int w2, int kk) {
  int w = blockIdx.x, i = blockIdx.y, agent = blockIdx.z;
  __shared__ float sx, sy;
  if (threadIdx.x == 0) {
    const float* R = rots + (size_t)(agent * 4 + kk) * 9;
    float X, Y;
    polar_xy(i, w, H, W, h1, w2, R[0], R[3], fxs[agent * 4 + kk], cxs[agent * 4 + kk], X, Y);
    sx = X; sy = Y;
  }
  __syncthreads();
  float x = fminf(fmaxf(sx, 0.f), (float)(W - 1));
  float y = fminf(fmaxf(sy, 0.f), (float)(H - 1));
  float x0f = floorf(x), y0f = floorf(y);
  float wx = x - x0f, wy = y - y0f;
  int x0 = (int)x0f, y0 = (int)y0f;
  int x1 = min(x0 + 1, W - 1), y1 = min(y0 + 1, H - 1);
  const float* ag = aWork + (size_t)agent * C * H * W;
  size_t outRow = (((size_t)agent * w2 + w) * h1 + i) * C;
  for (int c = threadIdx.x; c < C; c += blockDim.x) {
    const float* ch = ag + (size_t)c * H * W;
    float v00 = ch[y0 * W + x0], v01 = ch[y0 * W + x1];
    float v10 = ch[y1 * W + x0], v11 = ch[y1 * W + x1];
    float v = v00 * (1.f - wx) * (1.f - wy) + v01 * wx * (1.f - wy)
            + v10 * (1.f - wx) * wy + v11 * wx * wy;
    v += pa[i * C + c];
    aseq[outRow + c] = (_Float16)v;
  }
}

// b_seq: transpose b (n,4,c2,h2,w2)[kk] -> (n,w2,h2,c2) f16, + pb
__global__ void bseq_kernel(const float* __restrict__ b, const float* __restrict__ pb,
                            _Float16* __restrict__ bseq, int c2, int h2, int w2, int kk) {
  int w = blockIdx.x, j = blockIdx.y, agent = blockIdx.z;
  const float* bsrc = b + (((size_t)agent * 4 + kk) * c2) * h2 * w2;
  size_t outRow = (((size_t)agent * w2 + w) * h2 + j) * c2;
  for (int c = threadIdx.x; c < c2; c += blockDim.x) {
    float v = bsrc[((size_t)c * h2 + j) * w2 + w] + pb[j * c2 + c];
    bseq[outRow + c] = (_Float16)v;
  }
}

// ---------------------------------------------------------------------------
// WMMA GEMM: Out(MxN) = A(MxK,f16 row-major) @ Bfrag(KxN,f16 fragment-order)
// + bias. M % 128 == 0, N/K % 64 == 0. 128 threads = 4 waves; block tile
// 128x64; each wave owns 32 rows x 64 cols. Per k-step: preload 2 A-frags +
// 4 B-frags (batched b128 loads), then burst 8 WMMAs sharing the B-frags.
// ---------------------------------------------------------------------------
template <typename OutT>
__global__ void gemm_wmma_kernel(const _Float16* __restrict__ A,
                                 const _Float16* __restrict__ Bfrag,
                                 const float* __restrict__ bias,
                                 OutT* __restrict__ Out,
                                 int M, int N, int K) {
  const int tileM = blockIdx.y * 128;
  const int tileN = blockIdx.x * 64;
  const int wave = threadIdx.x >> 5;
  const int lane = threadIdx.x & 31;
  const int laneHi = lane >> 4, laneLo = lane & 15;
  const int nTiles = N >> 4;

  v8f acc[2][4];
#pragma unroll
  for (int s = 0; s < 2; ++s)
#pragma unroll
    for (int t = 0; t < 4; ++t)
#pragma unroll
      for (int r = 0; r < 8; ++r) acc[s][t][r] = 0.0f;

  const _Float16* Arow0 = A + (size_t)(tileM + (wave << 5) + laneLo) * K;
  const _Float16* Arow1 = Arow0 + (size_t)16 * K;
  for (int k0 = 0; k0 < K; k0 += 32) {
    // A 16x32 fragments: lane holds k [k0+8*hi,+8) and [k0+16+8*hi,+8)
    v8h a0lo = *(const v8h*)(Arow0 + k0 + laneHi * 8);
    v8h a0hi = *(const v8h*)(Arow0 + k0 + 16 + laneHi * 8);
    v8h a1lo = *(const v8h*)(Arow1 + k0 + laneHi * 8);
    v8h a1hi = *(const v8h*)(Arow1 + k0 + 16 + laneHi * 8);
    // B fragments: contiguous 32B per lane, preloaded as a batch
    const _Float16* Bk = Bfrag + ((size_t)(k0 >> 5) * nTiles) * 512;
    v16h bf[4];
#pragma unroll
    for (int t = 0; t < 4; ++t) {
      const _Float16* bp = Bk + (((size_t)((tileN >> 4) + t)) * 32 + lane) * 16;
      bf[t] = cat8(*(const v8h*)(bp), *(const v8h*)(bp + 8));
    }
    if (k0 + 32 < K) {  // global_prefetch_b8 for the next A chunk
      __builtin_prefetch(Arow0 + k0 + 32, 0, 1);
      __builtin_prefetch(Arow1 + k0 + 32, 0, 1);
    }
    v16h af0 = cat8(a0lo, a0hi);
    v16h af1 = cat8(a1lo, a1hi);
#pragma unroll
    for (int t = 0; t < 4; ++t)
      acc[0][t] = __builtin_amdgcn_wmma_f32_16x16x32_f16(
          false, af0, false, bf[t], (short)0, acc[0][t], false, false);
#pragma unroll
    for (int t = 0; t < 4; ++t)
      acc[1][t] = __builtin_amdgcn_wmma_f32_16x16x32_f16(
          false, af1, false, bf[t], (short)0, acc[1][t], false, false);
  }

  // C/D layout: VGPR r -> M = r + 8*laneHi; N = laneLo
#pragma unroll
  for (int s = 0; s < 2; ++s) {
    const int mBase = tileM + (wave << 5) + (s << 4) + (laneHi << 3);
#pragma unroll
    for (int t = 0; t < 4; ++t) {
      const int n = tileN + (t << 4) + laneLo;
      const float bv = bias ? bias[n] : 0.0f;
#pragma unroll
      for (int r = 0; r < 8; ++r)
        Out[(size_t)(mBase + r) * N + n] = (OutT)(acc[s][t][r] + bv);
    }
  }
}

// ---------------------------------------------------------------------------
// Attention: per (agent, w) block. K/V staged in LDS via async global->LDS;
// online softmax over h2 with head dim HD (8 or 16) in registers.
// ---------------------------------------------------------------------------
template <int HD>
__global__ void attention_kernel(const _Float16* __restrict__ qh,
                                 const _Float16* __restrict__ kh,
                                 const _Float16* __restrict__ vh,
                                 _Float16* __restrict__ attOut,
                                 int h1, int h2, int d, int w2) {
  extern __shared__ _Float16 smem[];  // [h2*d] K then [h2*d] V
  const int w = blockIdx.x, agent = blockIdx.y;
  const size_t kvBase = ((size_t)agent * w2 + w) * h2 * d;
  const int kvHalfs = h2 * d;
  {
    unsigned ldsK = (unsigned)(size_t)(void*)smem;           // LDS byte offset
    unsigned ldsV = ldsK + (unsigned)kvHalfs * 2u;
    const int chunks = kvHalfs >> 3;
    for (int cidx = threadIdx.x; cidx < chunks; cidx += blockDim.x) {
      async_g2l_b128(ldsK + (unsigned)cidx * 16u, kh + kvBase + (size_t)cidx * 8);
      async_g2l_b128(ldsV + (unsigned)cidx * 16u, vh + kvBase + (size_t)cidx * 8);
    }
    wait_async0();
  }
  __syncthreads();
  const size_t qBase = ((size_t)agent * w2 + w) * (size_t)h1 * d;
  for (int pair = threadIdx.x; pair < h1 * NHEADS; pair += blockDim.x) {
    const int i = pair >> 3, h = pair & 7;
    const _Float16* qrow = qh + qBase + (size_t)i * d + h * HD;
    float qv[HD];
#pragma unroll
    for (int t = 0; t < HD; ++t) qv[t] = (float)qrow[t];
    float mval = -1e30f, l = 0.f, oacc[HD];
#pragma unroll
    for (int t = 0; t < HD; ++t) oacc[t] = 0.f;
    for (int j = 0; j < h2; ++j) {
      const _Float16* krow = smem + j * d + h * HD;
      float s = 0.f;
#pragma unroll
      for (int t = 0; t < HD; ++t) s += qv[t] * (float)krow[t];
      float mnew = fmaxf(mval, s);
      float corr = __expf(mval - mnew);
      float p = __expf(s - mnew);
      l = l * corr + p;
      const _Float16* vrow = smem + kvHalfs + j * d + h * HD;
#pragma unroll
      for (int t = 0; t < HD; ++t) oacc[t] = oacc[t] * corr + p * (float)vrow[t];
      mval = mnew;
    }
    float inv = 1.f / l;
    _Float16* orow = attOut + qBase + (size_t)i * d + h * HD;
#pragma unroll
    for (int t = 0; t < HD; ++t) orow[t] = (_Float16)(oacc[t] * inv);
  }
}

// ---------------------------------------------------------------------------
// Restore: rounded polar scatter-add with counts (atomics), then normalize+add.
// ---------------------------------------------------------------------------
__global__ void scatter_kernel(const float* __restrict__ proj,  // (n,w2,h1,C) f32
                               const float* __restrict__ rots,
                               const float* __restrict__ fxs,
                               const float* __restrict__ cxs,
                               float* __restrict__ acc,   // (n,C,H,W)
                               float* __restrict__ cnt,   // (n,H,W)
                               int C, int H, int W, int h1, int w2, int kk) {
  int w = blockIdx.x, i = blockIdx.y, agent = blockIdx.z;
  __shared__ int sIdx;
  if (threadIdx.x == 0) {
    const float* R = rots + (size_t)(agent * 4 + kk) * 9;
    float X, Y;
    polar_xy(i, w, H, W, h1, w2, R[0], R[3], fxs[agent * 4 + kk], cxs[agent * 4 + kk], X, Y);
    int xi = (int)fminf(fmaxf(roundf(X), 0.f), (float)(W - 1));
    int yi = (int)fminf(fmaxf(roundf(Y), 0.f), (float)(H - 1));
    sIdx = yi * W + xi;
    atomicAdd(&cnt[(size_t)agent * H * W + sIdx], 1.0f);
  }
  __syncthreads();
  const int p = sIdx;
  const float* prow = proj + (((size_t)agent * w2 + w) * h1 + i) * C;
  float* accA = acc + (size_t)agent * C * H * W;
  for (int c = threadIdx.x; c < C; c += blockDim.x)
    atomicAdd(&accA[(size_t)c * H * W + p], prow[c]);
}

__global__ void norm_add_kernel(float* __restrict__ aWork, const float* __restrict__ acc,
                                const float* __restrict__ cnt, int C, int HW, size_t total) {
  size_t idx = (size_t)blockIdx.x * blockDim.x + threadIdx.x;
  size_t stride = (size_t)gridDim.x * blockDim.x;
  for (; idx < total; idx += stride) {
    size_t agent = idx / ((size_t)C * HW);
    size_t hw = idx % HW;
    float cv = cnt[agent * HW + hw];
    aWork[idx] += acc[idx] / fmaxf(cv, 1.0f);
  }
}

// ---------------------------------------------------------------------------
// Host orchestration
// ---------------------------------------------------------------------------
extern "C" void kernel_launch(void* const* d_in, const int* in_sizes, int n_in,
                              void* d_out, int out_size, void* d_ws, size_t ws_size,
                              hipStream_t stream) {
  (void)in_sizes; (void)n_in; (void)out_size;
  const float* a0  = (const float*)d_in[0];
  const float* a1  = (const float*)d_in[1];
  const float* b0  = (const float*)d_in[2];
  const float* b1  = (const float*)d_in[3];
  const float* rots = (const float*)d_in[5];
  const float* fxs  = (const float*)d_in[7];
  const float* cxs  = (const float*)d_in[8];
  float* out = (float*)d_out;
  char* ws = (char*)d_ws;

  const int n = 4;

  size_t off = 0;
  auto walloc = [&](size_t bytes) { size_t o = off; off = (off + bytes + 255) & ~(size_t)255; return o; };
  const size_t MAXE_AROWS = 8388608;  // max n*w2*h1*d elements
  const size_t MAXE_BROWS = 4194304;  // max n*w2*h2*c2 elements
  const size_t MAXE_KROWS = 4194304;  // max n*w2*h2*d elements
  const size_t MAXE_ACC   = 8388608;  // max n*C*H*W
  const size_t MAXE_CNT   = 131072;   // max n*H*W
  size_t oAseq = walloc(MAXE_AROWS * 2);
  size_t oBseq = walloc(MAXE_BROWS * 2);
  size_t oQh   = walloc(MAXE_AROWS * 2);
  size_t oKh   = walloc(MAXE_KROWS * 2);
  size_t oVh   = walloc(MAXE_KROWS * 2);
  size_t oAtt  = walloc(MAXE_AROWS * 2);
  size_t oProj = walloc(MAXE_AROWS * 4);
  size_t oAcc  = walloc(MAXE_ACC * 4);
  size_t oCnt  = walloc(MAXE_CNT * 4);
  size_t oWq = walloc(128 * 128 * 2), oWk = walloc(128 * 128 * 2);
  size_t oWv = walloc(128 * 128 * 2), oWo = walloc(128 * 128 * 2);
  size_t oBq = walloc(128 * 4), oBk = walloc(128 * 4), oBv = walloc(128 * 4), oBo = walloc(128 * 4);
  if (ws_size < off) return;

  _Float16* Aseq = (_Float16*)(ws + oAseq);
  _Float16* Bseq = (_Float16*)(ws + oBseq);
  _Float16* Qh   = (_Float16*)(ws + oQh);
  _Float16* Kh   = (_Float16*)(ws + oKh);
  _Float16* Vh   = (_Float16*)(ws + oVh);
  _Float16* Att  = (_Float16*)(ws + oAtt);
  float* Proj = (float*)(ws + oProj);
  float* Acc  = (float*)(ws + oAcc);
  float* Cnt  = (float*)(ws + oCnt);
  _Float16* WqF = (_Float16*)(ws + oWq); _Float16* WkF = (_Float16*)(ws + oWk);
  _Float16* WvF = (_Float16*)(ws + oWv); _Float16* WoF = (_Float16*)(ws + oWo);
  float* bQ = (float*)(ws + oBq); float* bK = (float*)(ws + oBk);
  float* bV = (float*)(ws + oBv); float* bO = (float*)(ws + oBo);

  struct Lvl { int C, H, W, h1, w2, h2, c2, d; const float* a; const float* b; int base; size_t outOff; };
  Lvl lv[2] = {
    { 64, 128, 256, 128, 256, 64,  64,  64, a0, b0,  9, 0 },
    { 128, 64, 128,  64, 128, 32, 128, 128, a1, b1, 21, (size_t)n * 64 * 128 * 256 },
  };

  for (int L = 0; L < 2; ++L) {
    const Lvl& v = lv[L];
    const float* pa   = (const float*)d_in[v.base + 0];
    const float* pb   = (const float*)d_in[v.base + 1];
    const float* wq   = (const float*)d_in[v.base + 2];
    const float* bq   = (const float*)d_in[v.base + 3];
    const float* wk   = (const float*)d_in[v.base + 4];
    const float* bk   = (const float*)d_in[v.base + 5];
    const float* wv   = (const float*)d_in[v.base + 6];
    const float* bv   = (const float*)d_in[v.base + 7];
    const float* win  = (const float*)d_in[v.base + 8];
    const float* binp = (const float*)d_in[v.base + 9];
    const float* wout = (const float*)d_in[v.base + 10];
    const float* bout = (const float*)d_in[v.base + 11];

    float* aWork = out + v.outOff;
    const size_t aElems = (size_t)n * v.C * v.H * v.W;
    copy_f32_kernel<<<dim3(2048), dim3(256), 0, stream>>>(v.a, aWork, aElems);

    {
      int maxElems = ((v.c2 > v.d ? v.c2 : v.d) * v.d);
      dim3 grid((maxElems + 255) / 256, 4);
      combine_weights_kernel<<<grid, 256, 0, stream>>>(
          wq, bq, wk, bk, wv, bv, win, binp, wout, bout,
          WqF, WkF, WvF, WoF, bQ, bK, bV, bO, v.d, v.c2);
    }

    const int Ma = n * v.w2 * v.h1;
    const int Mb = n * v.w2 * v.h2;
    for (int cam = 0; cam < 2; ++cam) {
      const int kk = (cam == 0) ? 0 : 3;

      extract_seq_kernel<<<dim3(v.w2, v.h1, n), 64, 0, stream>>>(
          aWork, rots, fxs, cxs, pa, Aseq, v.C, v.H, v.W, v.h1, v.w2, kk);
      bseq_kernel<<<dim3(v.w2, v.h2, n), 64, 0, stream>>>(
          v.b, pb, Bseq, v.c2, v.h2, v.w2, kk);

      gemm_wmma_kernel<_Float16><<<dim3(v.d / 64, Ma / 128), 128, 0, stream>>>(
          Aseq, WqF, bQ, Qh, Ma, v.d, v.d);
      gemm_wmma_kernel<_Float16><<<dim3(v.d / 64, Mb / 128), 128, 0, stream>>>(
          Bseq, WkF, bK, Kh, Mb, v.d, v.c2);
      gemm_wmma_kernel<_Float16><<<dim3(v.d / 64, Mb / 128), 128, 0, stream>>>(
          Bseq, WvF, bV, Vh, Mb, v.d, v.c2);

      size_t shm = (size_t)4 * v.h2 * v.d;  // K+V tiles in halves
      if (v.d / NHEADS == 8)
        attention_kernel<8><<<dim3(v.w2, n), 128, shm, stream>>>(
            Qh, Kh, Vh, Att, v.h1, v.h2, v.d, v.w2);
      else
        attention_kernel<16><<<dim3(v.w2, n), 128, shm, stream>>>(
            Qh, Kh, Vh, Att, v.h1, v.h2, v.d, v.w2);

      gemm_wmma_kernel<float><<<dim3(v.d / 64, Ma / 128), 128, 0, stream>>>(
          Att, WoF, bO, Proj, Ma, v.d, v.d);

      hipMemsetAsync(Acc, 0, aElems * sizeof(float), stream);
      hipMemsetAsync(Cnt, 0, (size_t)n * v.H * v.W * sizeof(float), stream);
      scatter_kernel<<<dim3(v.w2, v.h1, n), 64, 0, stream>>>(
          Proj, rots, fxs, cxs, Acc, Cnt, v.C, v.H, v.W, v.h1, v.w2, kk);
      norm_add_kernel<<<dim3(2048), dim3(256), 0, stream>>>(
          aWork, Acc, Cnt, v.C, v.H * v.W, aElems);
    }
  }
}
